// RelativeTransformerEncoderLayer_66760971649638
// MI455X (gfx1250) — compile-verified, matrix operand order checked by reading, not run on hardware
//
#include <hip/hip_runtime.h>
#include <math.h>

// ---------------------------------------------------------------------------
// Types for CDNA5 WMMA (wave32)
// ---------------------------------------------------------------------------
typedef __bf16 bf16_t;
typedef __attribute__((ext_vector_type(16))) __bf16 v16bf;
typedef __attribute__((ext_vector_type(2)))  __bf16 bf16x2;
typedef __attribute__((ext_vector_type(8)))  float  v8f;

// Problem constants (match reference)
constexpr int BSZ = 4;
constexpr int SEQ = 1024;
constexpr int DIM = 1024;
constexpr int NH  = 16;
constexpr int HDM = 64;
constexpr int FFD = 4096;
constexpr int D3  = 3 * DIM;

#define LDS_STRIDE 40   // 32 K-elements + 8 pad (keeps 16B alignment of rows)

// ---------------------------------------------------------------------------
// f32 -> bf16 convert kernel (n must be a multiple of 256)
// ---------------------------------------------------------------------------
__global__ __launch_bounds__(256) void cvt_f32_to_bf16(const float* __restrict__ in,
                                                       bf16_t* __restrict__ out, int n) {
    int i = blockIdx.x * 256 + threadIdx.x;
    if (i < n) out[i] = (bf16_t)in[i];
}

// ---------------------------------------------------------------------------
// Transpose V: qkv_bf [B*S, 3D] (v part at col offset 2D, head h at h*HD)
//   -> vT [B, H, HD, S]
// ---------------------------------------------------------------------------
__global__ __launch_bounds__(256) void transpose_v_kernel(const bf16_t* __restrict__ qkv,
                                                          bf16_t* __restrict__ vT) {
    size_t idx = (size_t)blockIdx.x * 256 + threadIdx.x;   // over B*H*HD*S
    int s = (int)(idx & (SEQ - 1));
    size_t rest = idx >> 10;
    int d = (int)(rest & (HDM - 1)); rest >>= 6;
    int h = (int)(rest & (NH - 1));
    int b = (int)(rest >> 4);
    vT[idx] = qkv[((size_t)(b * SEQ + s)) * D3 + 2 * DIM + h * HDM + d];
}

// ---------------------------------------------------------------------------
// Fragment gather from an LDS tile, per ISA 7.12.2 16-bit A/B layout.
// For lane l: row = base + (l&15), half = l>>4; VGPR v holds
// K = kbase(v) + half*8 + {0,1}, kbase = {0,2,4,6,16,18,20,22}.
// ---------------------------------------------------------------------------
__device__ inline v16bf load_frag(const bf16_t* __restrict__ smem, int baseRow, int lane) {
    const int r    = baseRow + (lane & 15);
    const int half = (lane >> 4) & 1;
    const bf16_t* p = smem + r * LDS_STRIDE + half * 8;
    v16bf f;
#pragma unroll
    for (int v = 0; v < 8; ++v) {
        const int k2 = (v < 4) ? (2 * v) : (16 + 2 * (v - 4));
        bf16x2 pr = *(const bf16x2*)(p + k2);
        f[2 * v]     = pr.x;
        f[2 * v + 1] = pr.y;
    }
    return f;
}

// ---------------------------------------------------------------------------
// Generic bf16 WMMA GEMM:  C[M,N] = alpha * A[M,K] @ B[N,K]^T  (+ bias[N])
// mode 0: store f32      mode 1: store bf16      mode 2: exact GELU -> bf16
//
// Block tile 128 x BN (BN = 128 or 64), 8 waves (4 row-groups x 2 col-groups),
// wave tile 32 x (BN/2). K stepped by 32 through double-buffered LDS filled
// with GLOBAL_LOAD_ASYNC_TO_LDS_B128 (ASYNCcnt pipeline): copies for stage
// s+1 overlap WMMA compute of stage s.
// Requires: M%128==0, N%BN==0, K%32==0, lda/ldb%8==0, 16B-aligned bases.
// ---------------------------------------------------------------------------
template <int BN>
__global__ __launch_bounds__(256) void gemm_bf16_wmma(
    const bf16_t* __restrict__ A, int lda,
    const bf16_t* __restrict__ Bm, int ldb,
    const float* __restrict__ bias,
    void* __restrict__ C, int ldc,
    int M, int N, int K, float alpha, int mode) {

    constexpr int CT = BN / 32;        // wave col tiles of 16 (4 or 2)
    constexpr int GB = BN / 64;        // B-tile async instrs per thread (2 or 1)
    constexpr int G  = 2 + GB;         // async instrs per thread per stage

    __shared__ __align__(16) bf16_t As[2 * 128 * LDS_STRIDE];
    __shared__ __align__(16) bf16_t Bs[2 * BN * LDS_STRIDE];

    const int tid  = threadIdx.x;
    const int lane = tid & 31;
    const int wave = tid >> 5;
    const int m_w  = (wave & 3) * 32;        // 4 wave-rows x 32
    const int n_w  = (wave >> 2) * (BN / 2); // 2 wave-cols x BN/2

    const int m0 = blockIdx.y * 128;
    const int n0 = blockIdx.x * BN;

    v8f acc[2][CT] = {};

    // Issue async copies (global -> LDS, 16B per lane per instruction).
    auto issueA = [&](int k0, int st) {
#pragma unroll
        for (int i = 0; i < 2; ++i) {
            const int u   = tid + i * 256;
            const int row = u >> 2;          // 4 x 16B units per 32-wide row
            const int ch  = u & 3;
            unsigned long long ga = (unsigned long long)(uintptr_t)
                (A + (size_t)(m0 + row) * lda + k0 + ch * 8);
            unsigned la = (unsigned)(uintptr_t)
                (As + st * (128 * LDS_STRIDE) + row * LDS_STRIDE + ch * 8);
            asm volatile("global_load_async_to_lds_b128 %0, %1, off"
                         :: "v"(la), "v"(ga) : "memory");
        }
    };
    auto issueB = [&](int k0, int st) {
#pragma unroll
        for (int i = 0; i < GB; ++i) {
            const int u   = tid + i * 256;
            const int row = u >> 2;
            const int ch  = u & 3;
            unsigned long long ga = (unsigned long long)(uintptr_t)
                (Bm + (size_t)(n0 + row) * ldb + k0 + ch * 8);
            unsigned la = (unsigned)(uintptr_t)
                (Bs + st * (BN * LDS_STRIDE) + row * LDS_STRIDE + ch * 8);
            asm volatile("global_load_async_to_lds_b128 %0, %1, off"
                         :: "v"(la), "v"(ga) : "memory");
        }
    };

    issueA(0, 0);
    issueB(0, 0);

    const int NS = K / 32;
    for (int s = 0; s < NS; ++s) {
        const int buf = s & 1;
        if (s + 1 < NS) {
            // Prefetch next stage into the other buffer, then wait for all
            // OLDER async copies (this stage's) to complete: async loads
            // complete in order, so cnt <= G leaves only the new ones.
            issueA((s + 1) * 32, buf ^ 1);
            issueB((s + 1) * 32, buf ^ 1);
            asm volatile("s_wait_asynccnt %0" :: "i"(G) : "memory");
        } else {
            asm volatile("s_wait_asynccnt 0" ::: "memory");
        }
        __syncthreads();   // all waves' copies for stage s are in LDS

        const bf16_t* Ab = As + buf * (128 * LDS_STRIDE);
        const bf16_t* Bb = Bs + buf * (BN * LDS_STRIDE);

        v16bf af[2], bfv[CT];
#pragma unroll
        for (int r = 0; r < 2; ++r)  af[r]  = load_frag(Ab, m_w + r * 16, lane);
#pragma unroll
        for (int c = 0; c < CT; ++c) bfv[c] = load_frag(Bb, n_w + c * 16, lane);

#pragma unroll
        for (int r = 0; r < 2; ++r)
#pragma unroll
            for (int c = 0; c < CT; ++c)
                acc[r][c] = __builtin_amdgcn_wmma_f32_16x16x32_bf16(
                    false, af[r], false, bfv[c], (short)0, acc[r][c], false, false);

        __syncthreads();   // all reads done before buf is refilled (stage s+2)
    }

    // Epilogue. D layout: lanes 0-15 -> cols, lanes 16-31 -> rows +8;
    // VGPR i -> row +i.
    float*  Cf = (float*)C;
    bf16_t* Cb = (bf16_t*)C;
#pragma unroll
    for (int r = 0; r < 2; ++r) {
#pragma unroll
        for (int c = 0; c < CT; ++c) {
            const int col  = n0 + n_w + c * 16 + (lane & 15);
            const int row0 = m0 + m_w + r * 16 + ((lane >> 4) & 1) * 8;
            const float bv = bias ? bias[col] : 0.0f;
#pragma unroll
            for (int i = 0; i < 8; ++i) {
                const int row = row0 + i;
                float v = acc[r][c][i] * alpha + bv;
                if (mode == 0) {
                    Cf[(size_t)row * ldc + col] = v;
                } else {
                    if (mode == 2)
                        v = 0.5f * v * (1.0f + erff(v * 0.70710678118654752f));
                    Cb[(size_t)row * ldc + col] = (bf16_t)v;
                }
            }
        }
    }
}

// ---------------------------------------------------------------------------
// Softmax over one row with additive bias: P = softmax(scores + bias) -> bf16
// ---------------------------------------------------------------------------
__global__ __launch_bounds__(256) void softmax_bias_kernel(const float* __restrict__ scores,
                                                           const float* __restrict__ bias,
                                                           bf16_t* __restrict__ P) {
    const int t   = blockIdx.x;
    const int tid = threadIdx.x;
    __shared__ float red[256];

    float vals[4];
    float m = -INFINITY;
#pragma unroll
    for (int i = 0; i < 4; ++i) {
        const int j = tid + i * 256;
        const float v = scores[(size_t)t * SEQ + j] + bias[(size_t)t * SEQ + j];
        vals[i] = v;
        m = fmaxf(m, v);
    }
    red[tid] = m;
    __syncthreads();
    for (int s = 128; s > 0; s >>= 1) {
        if (tid < s) red[tid] = fmaxf(red[tid], red[tid + s]);
        __syncthreads();
    }
    m = red[0];
    __syncthreads();

    float sum = 0.0f;
#pragma unroll
    for (int i = 0; i < 4; ++i) {
        vals[i] = expf(vals[i] - m);
        sum += vals[i];
    }
    red[tid] = sum;
    __syncthreads();
    for (int s = 128; s > 0; s >>= 1) {
        if (tid < s) red[tid] += red[tid + s];
        __syncthreads();
    }
    const float inv = 1.0f / red[0];
#pragma unroll
    for (int i = 0; i < 4; ++i)
        P[(size_t)t * SEQ + tid + i * 256] = (bf16_t)(vals[i] * inv);
}

// ---------------------------------------------------------------------------
// Fused residual + LayerNorm over rows of width DIM (=1024).
// ---------------------------------------------------------------------------
__global__ __launch_bounds__(256) void add_ln_kernel(const float* __restrict__ a,
                                                     const float* __restrict__ res,
                                                     const float* __restrict__ g,
                                                     const float* __restrict__ b,
                                                     float* __restrict__ outf,
                                                     bf16_t* __restrict__ outb) {
    const int row = blockIdx.x;
    const int tid = threadIdx.x;
    __shared__ float red[256];

    float loc[4];
    float s = 0.0f;
#pragma unroll
    for (int i = 0; i < 4; ++i) {
        const int j = tid + i * 256;
        const float v = a[(size_t)row * DIM + j] + res[(size_t)row * DIM + j];
        loc[i] = v;
        s += v;
    }
    red[tid] = s;
    __syncthreads();
    for (int st = 128; st > 0; st >>= 1) {
        if (tid < st) red[tid] += red[tid + st];
        __syncthreads();
    }
    const float mean = red[0] * (1.0f / DIM);
    __syncthreads();

    float vs = 0.0f;
#pragma unroll
    for (int i = 0; i < 4; ++i) {
        const float d = loc[i] - mean;
        vs += d * d;
    }
    red[tid] = vs;
    __syncthreads();
    for (int st = 128; st > 0; st >>= 1) {
        if (tid < st) red[tid] += red[tid + st];
        __syncthreads();
    }
    const float inv = rsqrtf(red[0] * (1.0f / DIM) + 1e-5f);

#pragma unroll
    for (int i = 0; i < 4; ++i) {
        const int j = tid + i * 256;
        const float y = (loc[i] - mean) * inv * g[j] + b[j];
        outf[(size_t)row * DIM + j] = y;
        if (outb) outb[(size_t)row * DIM + j] = (bf16_t)y;
    }
}

// ---------------------------------------------------------------------------
// Host orchestration
// ---------------------------------------------------------------------------
extern "C" void kernel_launch(void* const* d_in, const int* in_sizes, int n_in,
                              void* d_out, int out_size, void* d_ws, size_t ws_size,
                              hipStream_t stream) {
    (void)in_sizes; (void)n_in; (void)out_size; (void)ws_size;

    const float* src       = (const float*)d_in[0];
    const float* attn_bias = (const float*)d_in[1];
    const float* Wqkv      = (const float*)d_in[2];
    const float* bqkv      = (const float*)d_in[3];
    const float* Wo        = (const float*)d_in[4];
    const float* bo        = (const float*)d_in[5];
    const float* g1        = (const float*)d_in[6];
    const float* b1n       = (const float*)d_in[7];
    const float* g2        = (const float*)d_in[8];
    const float* b2n       = (const float*)d_in[9];
    const float* W1        = (const float*)d_in[10];
    const float* b1        = (const float*)d_in[11];
    const float* W2        = (const float*)d_in[12];
    const float* b2        = (const float*)d_in[13];
    float* out = (float*)d_out;

    char* ws = (char*)d_ws;
    size_t off = 0;
    auto alloc = [&](size_t bytes) -> char* {
        char* p = ws + off;
        off += (bytes + 255) & ~(size_t)255;
        return p;
    };

    const size_t NT = (size_t)BSZ * SEQ;            // 4096 tokens

    bf16_t* src_bf  = (bf16_t*)alloc(NT * DIM * 2);
    bf16_t* Wqkv_bf = (bf16_t*)alloc((size_t)D3 * DIM * 2);
    bf16_t* Wo_bf   = (bf16_t*)alloc((size_t)DIM * DIM * 2);
    bf16_t* W1_bf   = (bf16_t*)alloc((size_t)FFD * DIM * 2);
    bf16_t* W2_bf   = (bf16_t*)alloc((size_t)DIM * FFD * 2);
    bf16_t* qkv_bf  = (bf16_t*)alloc(NT * D3 * 2);
    bf16_t* vT_bf   = (bf16_t*)alloc((size_t)BSZ * NH * HDM * SEQ * 2);
    float*  scores  = (float*) alloc((size_t)SEQ * SEQ * 4);
    bf16_t* P_bf    = (bf16_t*)alloc((size_t)SEQ * SEQ * 2);
    bf16_t* ctx_bf  = (bf16_t*)alloc(NT * DIM * 2);
    float*  y_f     = (float*) alloc(NT * DIM * 4);
    float*  x_f     = (float*) alloc(NT * DIM * 4);
    bf16_t* x_bf    = (bf16_t*)alloc(NT * DIM * 2);
    bf16_t* h_bf    = (bf16_t*)alloc(NT * FFD * 2);
    float*  f_f     = (float*) alloc(NT * DIM * 4);

    auto cvt = [&](const float* in, bf16_t* o, size_t n) {
        cvt_f32_to_bf16<<<(unsigned)(n / 256), 256, 0, stream>>>(in, o, (int)n);
    };
    auto gemm128 = [&](const bf16_t* A, int lda, const bf16_t* Bm, int ldb,
                       const float* bias, void* C, int ldc,
                       int M, int N, int K, float alpha, int mode) {
        dim3 grid((unsigned)(N / 128), (unsigned)(M / 128));
        gemm_bf16_wmma<128><<<grid, 256, 0, stream>>>(A, lda, Bm, ldb, bias, C, ldc,
                                                      M, N, K, alpha, mode);
    };
    auto gemm64 = [&](const bf16_t* A, int lda, const bf16_t* Bm, int ldb,
                      const float* bias, void* C, int ldc,
                      int M, int N, int K, float alpha, int mode) {
        dim3 grid((unsigned)(N / 64), (unsigned)(M / 128));
        gemm_bf16_wmma<64><<<grid, 256, 0, stream>>>(A, lda, Bm, ldb, bias, C, ldc,
                                                     M, N, K, alpha, mode);
    };

    // 1) Convert inputs/weights to bf16
    cvt(src,  src_bf,  NT * DIM);
    cvt(Wqkv, Wqkv_bf, (size_t)D3 * DIM);
    cvt(Wo,   Wo_bf,   (size_t)DIM * DIM);
    cvt(W1,   W1_bf,   (size_t)FFD * DIM);
    cvt(W2,   W2_bf,   (size_t)DIM * FFD);

    // 2) QKV projection: qkv = src @ Wqkv^T + bqkv  -> bf16 [NT, 3D]
    gemm128(src_bf, DIM, Wqkv_bf, DIM, bqkv, qkv_bf, D3,
            (int)NT, D3, DIM, 1.0f, /*bf16*/1);

    // 3) Transpose V -> [B, H, HD, S]
    transpose_v_kernel<<<(unsigned)((size_t)BSZ * NH * HDM * SEQ / 256), 256, 0, stream>>>(
        qkv_bf, vT_bf);

    // 4) Attention per (b, h)
    const float inv_sqrt_hd = 0.125f;  // 1/sqrt(64)
    for (int b = 0; b < BSZ; ++b) {
        for (int h = 0; h < NH; ++h) {
            const bf16_t* qA = qkv_bf + (size_t)b * SEQ * D3 + h * HDM;
            const bf16_t* kB = qkv_bf + (size_t)b * SEQ * D3 + DIM + h * HDM;
            // scores = (q @ k^T) / sqrt(HD)   [S, S] f32
            gemm128(qA, D3, kB, D3, nullptr, scores, SEQ,
                    SEQ, SEQ, HDM, inv_sqrt_hd, /*f32*/0);
            // P = softmax(scores + attn_bias[h]) -> bf16
            softmax_bias_kernel<<<SEQ, 256, 0, stream>>>(
                scores, attn_bias + (size_t)h * SEQ * SEQ, P_bf);
            // ctx[:, h*HD:(h+1)*HD] = P @ V   (V as [HD, S])
            gemm64(P_bf, SEQ, vT_bf + ((size_t)(b * NH + h) * HDM) * SEQ, SEQ,
                   nullptr, ctx_bf + (size_t)b * SEQ * DIM + h * HDM, DIM,
                   SEQ, HDM, SEQ, 1.0f, /*bf16*/1);
        }
    }

    // 5) Output projection: y = ctx @ Wo^T + bo  (f32)
    gemm128(ctx_bf, DIM, Wo_bf, DIM, bo, y_f, DIM,
            (int)NT, DIM, DIM, 1.0f, /*f32*/0);

    // 6) x = LN(src + y) -> f32 + bf16
    add_ln_kernel<<<(unsigned)NT, 256, 0, stream>>>(y_f, src, g1, b1n, x_f, x_bf);

    // 7) FF1: h = gelu(x @ W1^T + b1) -> bf16 [NT, FF]
    gemm128(x_bf, DIM, W1_bf, DIM, b1, h_bf, FFD,
            (int)NT, FFD, DIM, 1.0f, /*gelu->bf16*/2);

    // 8) FF2: f = h @ W2^T + b2 -> f32
    gemm128(h_bf, FFD, W2_bf, FFD, b2, f_f, DIM,
            (int)NT, DIM, FFD, 1.0f, /*f32*/0);

    // 9) out = LN(x + f)
    add_ln_kernel<<<(unsigned)NT, 256, 0, stream>>>(f_f, x_f, g2, b2n, out, nullptr);
}
